// Net_85830626443707
// MI455X (gfx1250) — matrix operand order repaired and checked by reading, not compiled, not run on hardware
//
#include <hip/hip_runtime.h>
#include <math.h>

// ---------------------------------------------------------------------------
// DGCNN-ish EdgeConv net for MI455X (gfx1250, wave32, WMMA).
// All GEMMs use v_wmma_f32_16x16x32_bf16. BatchNorm (batch stats) handled by
// stats-pass / apply-pass recompute scheme to avoid E-sized activations.
// Fragment feeding is vectorized: B operands are pre-transposed so each lane's
// fragment is two aligned 16-byte chunks, k and k+16 (2x b128); A fragments are 2x b128 + shuffle.
// ---------------------------------------------------------------------------

typedef __bf16 bf16;
typedef bf16  v16bf __attribute__((ext_vector_type(16)));
typedef bf16  v8bf  __attribute__((ext_vector_type(8)));
typedef float v8f   __attribute__((ext_vector_type(8)));

#define NNODES 50000
#define NEDGES 800000
#define BN_EPS 1e-5f

static __device__ __forceinline__ v8f wmma_bf16(v16bf a, v16bf b, v8f c) {
  // (neg_a, A, neg_b, B, c_mod, C, reuse_a, reuse_b)
  return __builtin_amdgcn_wmma_f32_16x16x32_bf16(false, a, false, b, (short)0, c, false, false);
}

// A fragment: 16(M) x 32(K) bf16 from row-major src (leading dim ld, ld%32==0).
// Lane L: row = r0 + L%16; K chunks [k0+8*(L/16), +8) and [k0+16+8*(L/16), +8)
// -> two aligned 16-byte loads, concatenated.
static __device__ __forceinline__ v16bf load_frag_a(const bf16* __restrict__ src, int ld,
                                                    int r0, int k0, int lane, int maxrow) {
  int row = r0 + (lane & 15);
  row = row > maxrow ? maxrow : row;
  const bf16* p = src + (size_t)row * ld + k0 + 8 * (lane >> 4);
  v8bf lo = *(const v8bf*)p;
  v8bf hi = *(const v8bf*)(p + 16);
  return __builtin_shufflevector(lo, hi, 0, 1, 2, 3, 4, 5, 6, 7,
                                 8, 9, 10, 11, 12, 13, 14, 15);
}

// B fragment: 32(K) x 16(N) from TRANSPOSED weights Wt[C][Kpad] (Kpad%32==0).
// Lane L holds rows [k0+8*(L/16), +8) and [k0+16+8*(L/16), +8) of column n0+L%16
// -> two aligned 16-byte loads, concatenated (matches load_frag_a).
static __device__ __forceinline__ v16bf load_frag_bt(const bf16* __restrict__ Wt, int Kpad,
                                                     int k0, int n0, int lane) {
  const bf16* p = Wt + (size_t)(n0 + (lane & 15)) * Kpad + k0 + 8 * (lane >> 4);
  v8bf lo = *(const v8bf*)p;
  v8bf hi = *(const v8bf*)(p + 16);
  return __builtin_shufflevector(lo, hi, 0, 1, 2, 3, 4, 5, 6, 7,
                                 8, 9, 10, 11, 12, 13, 14, 15);
}

// ---------------------------------------------------------------------------
// Fused EdgeConv kernel. 128 edges per block, 8 waves; wave w owns rows
// [16w,16w+16) x all 64 output cols (4 accumulators -> 4 WMMAs per K-step).
// PASS 1: gather + L1 -> column sum/sumsq (stat1)
// PASS 2: gather + L1 + BN1 + L2        -> stat2
// PASS 3: gather + L1 + BN1 + L2 + BN2  -> atomic scatter-add to xout[dst]
// ---------------------------------------------------------------------------
template <int PASS>
__global__ __launch_bounds__(256)
void edge_conv_kernel(const float* __restrict__ xprev, int F, int Kp1,
                      const int* __restrict__ src, const int* __restrict__ dst,
                      const bf16* __restrict__ W1t, const float* __restrict__ b1,
                      float* __restrict__ stat1, const float* __restrict__ ab1,
                      const bf16* __restrict__ W2t, const float* __restrict__ b2,
                      float* __restrict__ stat2, const float* __restrict__ ab2,
                      float* __restrict__ xout)
{
  __shared__ bf16  feat[128 * 128];  // gathered [x_i, x_j - x_i], K-padded (32KB)
  __shared__ bf16  h1s[128 * 64];    // BN1 output = layer-2 input (16KB)
  __shared__ float s_sum[64];
  __shared__ float s_ssq[64];

  const int tid  = threadIdx.x;
  const int lane = tid & 31;
  const int wv   = tid >> 5;
  const int r0   = wv * 16;            // local row base for this wave
  const int e0   = blockIdx.x * 128;   // E divisible by 128

  if (tid < 64) { s_sum[tid] = 0.f; s_ssq[tid] = 0.f; }

  // gather edge features (node features live in L2)
  for (int idx = tid; idx < 128 * F; idx += 256) {
    int r = idx / F, f = idx - r * F;
    int e = e0 + r;
    int di = dst[e], si = src[e];
    float xd = xprev[(size_t)di * F + f];
    float xs = xprev[(size_t)si * F + f];
    feat[r * 128 + f]     = (bf16)xd;
    feat[r * 128 + F + f] = (bf16)(xs - xd);
  }
  const int pad = Kp1 - 2 * F;         // 24 for conv1, 0 otherwise
  for (int idx = tid; idx < 128 * pad; idx += 256) {
    int r = idx / pad, c = idx - r * pad;
    feat[r * 128 + 2 * F + c] = (bf16)0.f;
  }
  __syncthreads();

  // -------- layer 1: [128 x Kp1] @ [Kp1 x 64] --------
  v8f acc[4] = {};
  for (int k0 = 0; k0 < Kp1; k0 += 32) {
    v16bf a = load_frag_a(feat, 128, r0, k0, lane, 127);
#pragma unroll
    for (int t = 0; t < 4; ++t) {
      v16bf b = load_frag_bt(W1t, Kp1, k0, 16 * t, lane);
      acc[t] = wmma_bf16(a, b, acc[t]);
    }
  }
  const int kh = lane >> 4;
  const int nn = lane & 15;

  if (PASS == 1) {
#pragma unroll
    for (int t = 0; t < 4; ++t) {
      int c = 16 * t + nn;
      float bv = b1[c];
      float s = 0.f, q = 0.f;
#pragma unroll
      for (int r = 0; r < 8; ++r) {
        float y = fmaxf(acc[t][r] + bv, 0.f);
        s += y; q += y * y;
      }
      atomicAdd(&s_sum[c], s); atomicAdd(&s_ssq[c], q);
    }
    __syncthreads();
    if (tid < 64) {
      atomicAdd(&stat1[tid],      s_sum[tid]);
      atomicAdd(&stat1[64 + tid], s_ssq[tid]);
    }
    return;
  }

  // BN1 apply -> LDS (bf16) for layer-2 A
#pragma unroll
  for (int t = 0; t < 4; ++t) {
    int c = 16 * t + nn;
    float bv = b1[c], av = ab1[c], sv = ab1[64 + c];
#pragma unroll
    for (int r = 0; r < 8; ++r) {
      int row = r0 + r + 8 * kh;
      float y = fmaxf(acc[t][r] + bv, 0.f) * av + sv;
      h1s[row * 64 + c] = (bf16)y;
    }
  }
  __syncthreads();

  // -------- layer 2: [128 x 64] @ [64 x 64] --------
#pragma unroll
  for (int t = 0; t < 4; ++t)
#pragma unroll
    for (int r = 0; r < 8; ++r) acc[t][r] = 0.f;
#pragma unroll
  for (int k0 = 0; k0 < 64; k0 += 32) {
    v16bf a = load_frag_a(h1s, 64, r0, k0, lane, 127);
#pragma unroll
    for (int t = 0; t < 4; ++t) {
      v16bf b = load_frag_bt(W2t, 64, k0, 16 * t, lane);
      acc[t] = wmma_bf16(a, b, acc[t]);
    }
  }

  if (PASS == 2) {
#pragma unroll
    for (int t = 0; t < 4; ++t) {
      int c = 16 * t + nn;
      float bv = b2[c];
      float s = 0.f, q = 0.f;
#pragma unroll
      for (int r = 0; r < 8; ++r) {
        float y = fmaxf(acc[t][r] + bv, 0.f);
        s += y; q += y * y;
      }
      atomicAdd(&s_sum[c], s); atomicAdd(&s_ssq[c], q);
    }
    __syncthreads();
    if (tid < 64) {
      atomicAdd(&stat2[tid],      s_sum[tid]);
      atomicAdd(&stat2[64 + tid], s_ssq[tid]);
    }
    return;
  }

  // PASS 3: BN2 + segment_sum scatter (atomic f32 add at dst nodes)
  {
    int dsts[8];
#pragma unroll
    for (int r = 0; r < 8; ++r) dsts[r] = dst[e0 + r0 + 8 * kh + r];
#pragma unroll
    for (int t = 0; t < 4; ++t) {
      int c = 16 * t + nn;
      float bv = b2[c], av = ab2[c], sv = ab2[64 + c];
#pragma unroll
      for (int r = 0; r < 8; ++r) {
        float y = fmaxf(acc[t][r] + bv, 0.f) * av + sv;
        atomicAdd(&xout[(size_t)dsts[r] * 64 + c], y);
      }
    }
  }
}

// ---------------------------------------------------------------------------
// Node-level Linear+ReLU(+BN). Block = 128 rows x 64 cols, 8 waves; wave w
// owns rows [16w,16w+16) x 64 cols. PASS 1: stats; PASS 2: BN apply + store.
// ---------------------------------------------------------------------------
template <int PASS>
__global__ __launch_bounds__(256)
void node_mlp_kernel(const bf16* __restrict__ A, int K, int C,
                     const bf16* __restrict__ Wt, const float* __restrict__ bias,
                     float* __restrict__ stat, const float* __restrict__ ab,
                     bf16* __restrict__ out, int nrows)
{
  __shared__ float s_sum[64];
  __shared__ float s_ssq[64];
  const int tid  = threadIdx.x;
  const int lane = tid & 31;
  const int wv   = tid >> 5;
  const int r0    = blockIdx.x * 128 + wv * 16;
  const int c0blk = blockIdx.y * 64;

  if (PASS == 1) {
    if (tid < 64) { s_sum[tid] = 0.f; s_ssq[tid] = 0.f; }
    __syncthreads();
  }

  v8f acc[4] = {};
  for (int k0 = 0; k0 < K; k0 += 32) {
    v16bf a = load_frag_a(A, K, r0, k0, lane, nrows - 1);
#pragma unroll
    for (int t = 0; t < 4; ++t) {
      v16bf b = load_frag_bt(Wt, K, k0, c0blk + 16 * t, lane);
      acc[t] = wmma_bf16(a, b, acc[t]);
    }
  }
  const int kh = lane >> 4;
  const int nn = lane & 15;

  if (PASS == 1) {
#pragma unroll
    for (int t = 0; t < 4; ++t) {
      int lc = 16 * t + nn;
      float bv = bias[c0blk + lc];
      float s = 0.f, q = 0.f;
#pragma unroll
      for (int r = 0; r < 8; ++r) {
        int row = r0 + r + 8 * kh;
        if (row < nrows) {
          float y = fmaxf(acc[t][r] + bv, 0.f);
          s += y; q += y * y;
        }
      }
      atomicAdd(&s_sum[lc], s); atomicAdd(&s_ssq[lc], q);
    }
    __syncthreads();
    if (tid < 64) {
      atomicAdd(&stat[c0blk + tid],     s_sum[tid]);
      atomicAdd(&stat[C + c0blk + tid], s_ssq[tid]);
    }
  } else {
#pragma unroll
    for (int t = 0; t < 4; ++t) {
      int c = c0blk + 16 * t + nn;
      float bv = bias[c], av = ab[c], sv = ab[C + c];
#pragma unroll
      for (int r = 0; r < 8; ++r) {
        int row = r0 + r + 8 * kh;
        if (row < nrows) {
          float y = fmaxf(acc[t][r] + bv, 0.f) * av + sv;
          out[(size_t)row * C + c] = (bf16)y;
        }
      }
    }
  }
}

// --------------------------- small helper kernels ---------------------------

__global__ void zero_f32_kernel(float* __restrict__ p, size_t n) {
  size_t i = (size_t)blockIdx.x * blockDim.x + threadIdx.x;
  size_t s = (size_t)gridDim.x * blockDim.x;
  for (; i < n; i += s) p[i] = 0.f;
}

// Transpose + K-pad weights: Wt[c*Kpad + k] = (k<K) ? W[k*C+c] : 0  (bf16)
__global__ void cvt_weight_t_kernel(const float* __restrict__ W, bf16* __restrict__ Wt,
                                    int K, int Kpad, int C) {
  size_t n = (size_t)Kpad * C;
  size_t i = (size_t)blockIdx.x * blockDim.x + threadIdx.x;
  size_t s = (size_t)gridDim.x * blockDim.x;
  for (; i < n; i += s) {
    int c = (int)(i / Kpad), k = (int)(i - (size_t)c * Kpad);
    Wt[i] = (k < K) ? (bf16)W[(size_t)k * C + c] : (bf16)0.f;
  }
}

// BN batch stats -> per-column scale/shift: y*a + s with a=g*rsqrt(var+eps)
__global__ void finalize_bn_kernel(const float* __restrict__ stat, int C, float inv_count,
                                   const float* __restrict__ g, const float* __restrict__ be,
                                   float* __restrict__ ab) {
  int c = blockIdx.x * blockDim.x + threadIdx.x;
  if (c >= C) return;
  float mu  = stat[c] * inv_count;
  float var = stat[C + c] * inv_count - mu * mu;
  float a   = g[c] * rsqrtf(var + BN_EPS);
  ab[c]     = a;
  ab[C + c] = be[c] - mu * a;
}

__global__ void concat3_kernel(const float* __restrict__ x1, const float* __restrict__ x2,
                               const float* __restrict__ x3, bf16* __restrict__ out) {
  size_t n = (size_t)NNODES * 64;
  size_t i = (size_t)blockIdx.x * blockDim.x + threadIdx.x;
  size_t s = (size_t)gridDim.x * blockDim.x;
  for (; i < n; i += s) {
    size_t r = i >> 6; int c = (int)(i & 63);
    out[r * 192 + c]       = (bf16)x1[i];
    out[r * 192 + 64 + c]  = (bf16)x2[i];
    out[r * 192 + 128 + c] = (bf16)x3[i];
  }
}

__global__ void out_logsoftmax_kernel(const bf16* __restrict__ h, const float* __restrict__ Wo,
                                      const float* __restrict__ bo, float* __restrict__ out) {
  int i = blockIdx.x * blockDim.x + threadIdx.x;
  if (i >= NNODES) return;
  const v8bf* hp = (const v8bf*)(h + (size_t)i * 128);
  float logits[16];
#pragma unroll
  for (int o = 0; o < 16; ++o) logits[o] = bo[o];
  for (int f8 = 0; f8 < 16; ++f8) {
    v8bf hv = hp[f8];
#pragma unroll
    for (int j = 0; j < 8; ++j) {
      float v = (float)hv[j];
      int f = f8 * 8 + j;
#pragma unroll
      for (int o = 0; o < 16; ++o) logits[o] += v * Wo[f * 16 + o];
    }
  }
  float mx = logits[0];
#pragma unroll
  for (int o = 1; o < 16; ++o) mx = fmaxf(mx, logits[o]);
  float se = 0.f;
#pragma unroll
  for (int o = 0; o < 16; ++o) se += __expf(logits[o] - mx);
  float lse = mx + __logf(se);
#pragma unroll
  for (int o = 0; o < 16; ++o) out[(size_t)i * 16 + o] = logits[o] - lse;
}

// ---------------------------------------------------------------------------
// Host orchestration.
// Input order assumption: d_in[0]=x, d_in[1]=edge_index, then params in JAX
// pytree order (dict keys sorted): per MLP layer {be, g, W, b};
// conv1(L0,L1), conv2, conv3, head1, head2, lin1, out{W,b}.
// ---------------------------------------------------------------------------
struct Lay { const float *be, *g, *W, *b; };

extern "C" void kernel_launch(void* const* d_in, const int* in_sizes, int n_in,
                              void* d_out, int out_size, void* d_ws, size_t ws_size,
                              hipStream_t stream)
{
  (void)in_sizes; (void)n_in; (void)out_size; (void)ws_size;
  const float* x  = (const float*)d_in[0];
  const int*   ei = (const int*)d_in[1];
  const int* src = ei;
  const int* dst = ei + NEDGES;

  auto L = [&](int base) {
    Lay l; l.be = (const float*)d_in[base]; l.g = (const float*)d_in[base + 1];
    l.W = (const float*)d_in[base + 2]; l.b = (const float*)d_in[base + 3]; return l;
  };
  Lay c1a = L(2),  c1b = L(6);
  Lay c2a = L(10), c2b = L(14);
  Lay c3a = L(18), c3b = L(22);
  Lay hd1 = L(26), hd2 = L(30), ln1 = L(34);
  const float* Wout = (const float*)d_in[38];
  const float* bout = (const float*)d_in[39];

  // ---- workspace carve-up (256B aligned) ----
  char* ws = (char*)d_ws;
  size_t off = 0;
  auto alloc = [&](size_t bytes) -> void* {
    void* p = ws + off; off += (bytes + 255) & ~(size_t)255; return p;
  };
  float* x1    = (float*)alloc((size_t)NNODES * 64 * 4);
  float* x2    = (float*)alloc((size_t)NNODES * 64 * 4);
  float* x3    = (float*)alloc((size_t)NNODES * 64 * 4);
  float* stats = (float*)alloc(3584 * 4);              // end of zeroed region
  size_t zero_n = (size_t)(((char*)stats + 3584 * 4) - (char*)x1) / 4;
  float* abz   = (float*)alloc(3584 * 4);

  float *st_c1a = stats,        *st_c1b = stats + 128;
  float *st_c2a = stats + 256,  *st_c2b = stats + 384;
  float *st_c3a = stats + 512,  *st_c3b = stats + 640;
  float *st_l1  = stats + 768,  *st_h1  = stats + 2816, *st_h2 = stats + 3328;
  float *ab_c1a = abz,          *ab_c1b = abz + 128;
  float *ab_c2a = abz + 256,    *ab_c2b = abz + 384;
  float *ab_c3a = abz + 512,    *ab_c3b = abz + 640;
  float *ab_l1  = abz + 768,    *ab_h1  = abz + 2816,   *ab_h2 = abz + 3328;

  bf16* x123 = (bf16*)alloc((size_t)NNODES * 192 * 2);
  bf16* hb1  = (bf16*)alloc((size_t)NNODES * 1024 * 2);
  bf16* hb2  = (bf16*)alloc((size_t)NNODES * 256 * 2);
  bf16* hb3  = (bf16*)alloc((size_t)NNODES * 128 * 2);
  // transposed bf16 weights: Wt[C][Kpad]
  bf16* W1a  = (bf16*)alloc((size_t)64 * 32 * 2);
  bf16* W1b  = (bf16*)alloc((size_t)64 * 64 * 2);
  bf16* W2a  = (bf16*)alloc((size_t)64 * 128 * 2);
  bf16* W2b  = (bf16*)alloc((size_t)64 * 64 * 2);
  bf16* W3a  = (bf16*)alloc((size_t)64 * 128 * 2);
  bf16* W3b  = (bf16*)alloc((size_t)64 * 64 * 2);
  bf16* Wl1  = (bf16*)alloc((size_t)1024 * 192 * 2);
  bf16* Wh1  = (bf16*)alloc((size_t)256 * 1024 * 2);
  bf16* Wh2  = (bf16*)alloc((size_t)128 * 256 * 2);

  const float invE = 1.f / (float)NEDGES;
  const float invN = 1.f / (float)NNODES;
  const int EB = NEDGES / 128;                 // 6250 edge blocks

  zero_f32_kernel<<<4096, 256, 0, stream>>>(x1, zero_n);
  cvt_weight_t_kernel<<<64, 256, 0, stream>>>(c1a.W, W1a, 8, 32, 64);
  cvt_weight_t_kernel<<<64, 256, 0, stream>>>(c1b.W, W1b, 64, 64, 64);
  cvt_weight_t_kernel<<<64, 256, 0, stream>>>(c2a.W, W2a, 128, 128, 64);
  cvt_weight_t_kernel<<<64, 256, 0, stream>>>(c2b.W, W2b, 64, 64, 64);
  cvt_weight_t_kernel<<<64, 256, 0, stream>>>(c3a.W, W3a, 128, 128, 64);
  cvt_weight_t_kernel<<<64, 256, 0, stream>>>(c3b.W, W3b, 64, 64, 64);
  cvt_weight_t_kernel<<<256, 256, 0, stream>>>(ln1.W, Wl1, 192, 192, 1024);
  cvt_weight_t_kernel<<<256, 256, 0, stream>>>(hd1.W, Wh1, 1024, 1024, 256);
  cvt_weight_t_kernel<<<128, 256, 0, stream>>>(hd2.W, Wh2, 256, 256, 128);

  // ---- conv1: x[N,4] -> x1[N,64] ----
  edge_conv_kernel<1><<<EB, 256, 0, stream>>>(x, 4, 32, src, dst, W1a, c1a.b, st_c1a, nullptr, W1b, c1b.b, nullptr, nullptr, nullptr);
  finalize_bn_kernel<<<1, 64, 0, stream>>>(st_c1a, 64, invE, c1a.g, c1a.be, ab_c1a);
  edge_conv_kernel<2><<<EB, 256, 0, stream>>>(x, 4, 32, src, dst, W1a, c1a.b, nullptr, ab_c1a, W1b, c1b.b, st_c1b, nullptr, nullptr);
  finalize_bn_kernel<<<1, 64, 0, stream>>>(st_c1b, 64, invE, c1b.g, c1b.be, ab_c1b);
  edge_conv_kernel<3><<<EB, 256, 0, stream>>>(x, 4, 32, src, dst, W1a, c1a.b, nullptr, ab_c1a, W1b, c1b.b, nullptr, ab_c1b, x1);

  // ---- conv2: x1 -> x2 ----
  edge_conv_kernel<1><<<EB, 256, 0, stream>>>(x1, 64, 128, src, dst, W2a, c2a.b, st_c2a, nullptr, W2b, c2b.b, nullptr, nullptr, nullptr);
  finalize_bn_kernel<<<1, 64, 0, stream>>>(st_c2a, 64, invE, c2a.g, c2a.be, ab_c2a);
  edge_conv_kernel<2><<<EB, 256, 0, stream>>>(x1, 64, 128, src, dst, W2a, c2a.b, nullptr, ab_c2a, W2b, c2b.b, st_c2b, nullptr, nullptr);
  finalize_bn_kernel<<<1, 64, 0, stream>>>(st_c2b, 64, invE, c2b.g, c2b.be, ab_c2b);
  edge_conv_kernel<3><<<EB, 256, 0, stream>>>(x1, 64, 128, src, dst, W2a, c2a.b, nullptr, ab_c2a, W2b, c2b.b, nullptr, ab_c2b, x2);

  // ---- conv3: x2 -> x3 ----
  edge_conv_kernel<1><<<EB, 256, 0, stream>>>(x2, 64, 128, src, dst, W3a, c3a.b, st_c3a, nullptr, W3b, c3b.b, nullptr, nullptr, nullptr);
  finalize_bn_kernel<<<1, 64, 0, stream>>>(st_c3a, 64, invE, c3a.g, c3a.be, ab_c3a);
  edge_conv_kernel<2><<<EB, 256, 0, stream>>>(x2, 64, 128, src, dst, W3a, c3a.b, nullptr, ab_c3a, W3b, c3b.b, st_c3b, nullptr, nullptr);
  finalize_bn_kernel<<<1, 64, 0, stream>>>(st_c3b, 64, invE, c3b.g, c3b.be, ab_c3b);
  edge_conv_kernel<3><<<EB, 256, 0, stream>>>(x2, 64, 128, src, dst, W3a, c3a.b, nullptr, ab_c3a, W3b, c3b.b, nullptr, ab_c3b, x3);

  // ---- concat -> lin1 -> head1 -> head2 ----
  concat3_kernel<<<4096, 256, 0, stream>>>(x1, x2, x3, x123);

  const int NB = (NNODES + 127) / 128;         // 391 node-row blocks
  dim3 g1(NB, 1024 / 64);
  node_mlp_kernel<1><<<g1, 256, 0, stream>>>(x123, 192, 1024, Wl1, ln1.b, st_l1, nullptr, nullptr, NNODES);
  finalize_bn_kernel<<<8, 128, 0, stream>>>(st_l1, 1024, invN, ln1.g, ln1.be, ab_l1);
  node_mlp_kernel<2><<<g1, 256, 0, stream>>>(x123, 192, 1024, Wl1, ln1.b, nullptr, ab_l1, hb1, NNODES);

  dim3 g2(NB, 256 / 64);
  node_mlp_kernel<1><<<g2, 256, 0, stream>>>(hb1, 1024, 256, Wh1, hd1.b, st_h1, nullptr, nullptr, NNODES);
  finalize_bn_kernel<<<2, 128, 0, stream>>>(st_h1, 256, invN, hd1.g, hd1.be, ab_h1);
  node_mlp_kernel<2><<<g2, 256, 0, stream>>>(hb1, 1024, 256, Wh1, hd1.b, nullptr, ab_h1, hb2, NNODES);

  dim3 g3(NB, 128 / 64);
  node_mlp_kernel<1><<<g3, 256, 0, stream>>>(hb2, 256, 128, Wh2, hd2.b, st_h2, nullptr, nullptr, NNODES);
  finalize_bn_kernel<<<1, 128, 0, stream>>>(st_h2, 128, invN, hd2.g, hd2.be, ab_h2);
  node_mlp_kernel<2><<<g3, 256, 0, stream>>>(hb2, 256, 128, Wh2, hd2.b, nullptr, ab_h2, hb3, NNODES);

  // ---- final linear + log_softmax ----
  out_logsoftmax_kernel<<<(NNODES + 255) / 256, 256, 0, stream>>>(hb3, Wout, bout, (float*)d_out);
}
